// MultiheadedAttention_85057532330058
// MI455X (gfx1250) — matrix operand order repaired and checked by reading, not compiled, not run on hardware
//
#include <hip/hip_runtime.h>
#include <hip/hip_bf16.h>
#include <stdint.h>

// ---------------------------------------------------------------------------
// MI455X (gfx1250) fused multi-head attention, wave32 + v_wmma_f32_16x16x32_f16
// All operands pre-converted to f16 once (L2-resident); GEMM inner loops are
// pure {b128 load clauses -> v_wmma} with no conversion VALU.
// ---------------------------------------------------------------------------

typedef _Float16 half_t;
typedef __attribute__((ext_vector_type(16))) _Float16 v16h;
typedef __attribute__((ext_vector_type(8)))  _Float16 v8h;
typedef __attribute__((ext_vector_type(4)))  _Float16 v4h;
typedef __attribute__((ext_vector_type(8)))  float    v8f;
typedef __attribute__((ext_vector_type(4)))  float    v4f;
typedef uint32_t u32;

static constexpr int B_    = 2;
static constexpr int T_    = 2048;
static constexpr int H_    = 16;
static constexpr int DK_   = 64;
static constexpr int D_    = 1024;
static constexpr int NTOK_ = B_ * T_;      // 4096

#define WMMA(A, Bf, C) \
  __builtin_amdgcn_wmma_f32_16x16x32_f16(false, (A), false, (Bf), (short)0, (C), false, false)

__device__ __forceinline__ u32 pack2h(float a, float b) {
  union { _Float16 h[2]; u32 u; } cv;
  cv.h[0] = (_Float16)a; cv.h[1] = (_Float16)b;
  return cv.u;
}

__device__ __forceinline__ v16h cat8(v8h lo, v8h hi) {
  v16h r;
#pragma unroll
  for (int i = 0; i < 8; ++i) { r[i] = lo[i]; r[i + 8] = hi[i]; }
  return r;
}

// ---------------------------------------------------------------------------
// fp32 -> f16 conversion (weights and activations; one pass, then L2-resident)
// ---------------------------------------------------------------------------
__global__ __launch_bounds__(256) void cvt_f32_f16(const float* __restrict__ in,
                                                   half_t* __restrict__ out, int n4) {
  int i = blockIdx.x * blockDim.x + threadIdx.x;
  if (i >= n4) return;
  v4f x = ((const v4f*)in)[i];
  v4h o;
#pragma unroll
  for (int j = 0; j < 4; ++j) o[j] = (half_t)x[j];
  ((v4h*)out)[i] = o;
}

// ---------------------------------------------------------------------------
// Projection, feature-major (M = out-features, N = tokens): Y = X @ W^T + b,
// stored (B, H, T, DK) f16. All fragment loads contiguous b128; per K-step:
// one 2-load A-frag + one 8-load B-frag clause + 4 WMMA.
// One wave: 16 features x 64 tokens. 4096 waves.
// ---------------------------------------------------------------------------
__global__ __launch_bounds__(256) void proj_featmajor(
    const half_t* __restrict__ Xh,   // (NTOK, D) f16
    const half_t* __restrict__ Wh,   // (D, D) f16, row = output feature
    const float* __restrict__ bias,  // (D)
    half_t* __restrict__ Out)        // (B, H, T, DK) f16
{
  const int lane = threadIdx.x & 31;
  const int col  = lane & 15;
  const int hi   = lane >> 4;
  const int w    = blockIdx.x * 8 + (threadIdx.x >> 5);
  const int fbase   = (w & 63) * 16;   // feature tile
  const int tokbase = (w >> 6) * 64;   // token strip

  v8f acc[4];
#pragma unroll
  for (int nt = 0; nt < 4; ++nt)
#pragma unroll
    for (int j = 0; j < 8; ++j) acc[nt][j] = 0.f;

  const half_t* wrow = Wh + (size_t)(fbase + col) * D_;
  const half_t* xrow[4];
#pragma unroll
  for (int nt = 0; nt < 4; ++nt)
    xrow[nt] = Xh + (size_t)(tokbase + nt * 16 + col) * D_ + 16 * hi;

  for (int kb = 0; kb < D_; kb += 32) {
    // A-frag: lane row = feature, K = 16*hi + {0..15} (contiguous)
    v8h a0 = *(const v8h*)(wrow + kb + 16 * hi);
    v8h a1 = *(const v8h*)(wrow + kb + 16 * hi + 8);
    // B-frags: lane col = token, K = 16*hi + i (16 contiguous f16)
    v8h b0[4], b1[4];
#pragma unroll
    for (int nt = 0; nt < 4; ++nt) {
      b0[nt] = ((const v8h*)(xrow[nt] + kb))[0];
      b1[nt] = ((const v8h*)(xrow[nt] + kb))[1];
    }
    v16h a = cat8(a0, a1);
#pragma unroll
    for (int nt = 0; nt < 4; ++nt)
      acc[nt] = WMMA(a, cat8(b0[nt], b1[nt]), acc[nt]);
  }
  // epilogue: C rows = feature fbase + r + 8*hi, cols = token
  const int h  = fbase >> 6;
  const int d0 = (fbase & 63) + 8 * hi;
  float bv[8];
#pragma unroll
  for (int r = 0; r < 8; ++r) bv[r] = bias[fbase + 8 * hi + r];
#pragma unroll
  for (int nt = 0; nt < 4; ++nt) {
    int tok = tokbase + nt * 16 + col;
    int bb = tok / T_, tt = tok % T_;
    half_t* dst = Out + (((size_t)bb * H_ + h) * T_ + tt) * DK_ + d0;
    v8h o;
#pragma unroll
    for (int r = 0; r < 8; ++r) o[r] = (half_t)(acc[nt][r] + bv[r]);
    *(v8h*)dst = o;
  }
}

// ---------------------------------------------------------------------------
// V projection, token-major (M = tokens, N = features), storing V transposed
// as (B, H, DK, T) f16 so attention's V^T A-fragments are contiguous.
// One wave: 16 tokens x 64 features. 4096 waves.
// ---------------------------------------------------------------------------
__global__ __launch_bounds__(256) void proj_v_transposed(
    const half_t* __restrict__ Xh,   // (NTOK, D) f16
    const half_t* __restrict__ Wh,   // (D, D) f16
    const float* __restrict__ bias,  // (D)
    half_t* __restrict__ Vt)         // (B, H, DK, T) f16
{
  const int lane = threadIdx.x & 31;
  const int col  = lane & 15;
  const int hi   = lane >> 4;
  const int w    = blockIdx.x * 8 + (threadIdx.x >> 5);
  const int fbase0  = (w & 15) * 64;   // feature strip
  const int tokbase = (w >> 4) * 16;   // token tile

  v8f acc[4];
#pragma unroll
  for (int nt = 0; nt < 4; ++nt)
#pragma unroll
    for (int j = 0; j < 8; ++j) acc[nt][j] = 0.f;

  const half_t* xrow = Xh + (size_t)(tokbase + col) * D_;
  const half_t* wr[4];
#pragma unroll
  for (int nt = 0; nt < 4; ++nt)
    wr[nt] = Wh + (size_t)(fbase0 + nt * 16 + col) * D_ + 16 * hi;

  for (int kb = 0; kb < D_; kb += 32) {
    // A-frag: lane row = token, K = 16*hi + {0..15} (contiguous)
    v8h a0 = *(const v8h*)(xrow + kb + 16 * hi);
    v8h a1 = *(const v8h*)(xrow + kb + 16 * hi + 8);
    // B-frags: lane col = feature, B[k][n] = W[feature][k] contiguous f16
    v8h b0[4], b1[4];
#pragma unroll
    for (int nt = 0; nt < 4; ++nt) {
      b0[nt] = ((const v8h*)(wr[nt] + kb))[0];
      b1[nt] = ((const v8h*)(wr[nt] + kb))[1];
    }
    v16h a = cat8(a0, a1);
#pragma unroll
    for (int nt = 0; nt < 4; ++nt)
      acc[nt] = WMMA(a, cat8(b0[nt], b1[nt]), acc[nt]);
  }
  // epilogue: C rows = token tokbase + r + 8*hi, cols = feature
  const int bb  = tokbase / T_;
  const int tt0 = (tokbase % T_) + 8 * hi;
#pragma unroll
  for (int nt = 0; nt < 4; ++nt) {
    int feat = fbase0 + nt * 16 + col;
    float bvv = bias[feat];
    half_t* dst = Vt + (((size_t)bb * H_ + (feat >> 6)) * DK_ + (feat & 63)) * T_ + tt0;
    v8h o;
#pragma unroll
    for (int r = 0; r < 8; ++r) o[r] = (half_t)(acc[nt][r] + bvv);
    *(v8h*)dst = o;
  }
}

// ---------------------------------------------------------------------------
// Flash attention (no 1/sqrt(dk) scale, matching reference), one wave per
// 16-query tile of one (b,h). Computes S^T = K·Q^T so softmax is lane-local;
// accumulates out^T = V^T · P^T so the online rescale is shuffle-free.
// ---------------------------------------------------------------------------
__global__ __launch_bounds__(256) void flash_attn(
    const half_t* __restrict__ Qh,   // (B,H,T,DK) f16
    const half_t* __restrict__ Kh,   // (B,H,T,DK) f16
    const half_t* __restrict__ Vt,   // (B,H,DK,T) f16
    const unsigned char* __restrict__ mask, // (B,T,T) bool
    half_t* __restrict__ Xo)         // (NTOK, D) f16
{
  const int lane = threadIdx.x & 31;
  const int col  = lane & 15;
  const int hi   = lane >> 4;
  const int w    = blockIdx.x * 8 + (threadIdx.x >> 5);
  const int qbase = (w & 127) * 16;
  const int bh    = w >> 7;

  const half_t* Qp = Qh + (size_t)bh * T_ * DK_;
  const half_t* Kp = Kh + (size_t)bh * T_ * DK_;
  const half_t* Vp = Vt + (size_t)bh * DK_ * T_;
  const unsigned char* mp = mask + (size_t)(bh >> 4) * T_ * T_ + (size_t)(qbase + col) * T_;

  // Q B-fragments (lane col = query), d-blocks [0,32) and [32,64)
  v16h bq0, bq1;
  {
    const half_t* qr = Qp + (size_t)(qbase + col) * DK_ + 8 * hi;
    bq0 = cat8(((const v8h*)qr)[0], ((const v8h*)(qr + 16))[0]);
    bq1 = cat8(((const v8h*)(qr + 32))[0], ((const v8h*)(qr + 48))[0]);
  }

  float m = -1e30f, l = 0.f;
  v8f acc[4];
#pragma unroll
  for (int f = 0; f < 4; ++f)
#pragma unroll
    for (int j = 0; j < 8; ++j) acc[f][j] = 0.f;

  for (int kb = 0; kb < T_; kb += 32) {
    // ---- S^T = K-tile · Q^T ------------------------------------------------
    v8h kf[2][4];
    uint2 mb[2];
#pragma unroll
    for (int sub = 0; sub < 2; ++sub) {
      const half_t* kr = Kp + (size_t)(kb + sub * 16 + col) * DK_ + 8 * hi;
#pragma unroll
      for (int p = 0; p < 4; ++p) kf[sub][p] = ((const v8h*)(kr + 16 * p))[0];
      mb[sub] = *(const uint2*)(mp + kb + sub * 16 + 8 * hi);
    }
    float sv[2][8];
#pragma unroll
    for (int sub = 0; sub < 2; ++sub) {
      v8f z;
#pragma unroll
      for (int j = 0; j < 8; ++j) z[j] = 0.f;
      v8f st = WMMA(cat8(kf[sub][0], kf[sub][1]), bq0, z);
      st = WMMA(cat8(kf[sub][2], kf[sub][3]), bq1, st);
#pragma unroll
      for (int j = 0; j < 8; ++j) {
        unsigned byte =
            (j < 4 ? (mb[sub].x >> (8 * j)) : (mb[sub].y >> (8 * (j - 4)))) & 0xFFu;
        sv[sub][j] = byte ? st[j] : -1e9f;
      }
    }
    // ---- online softmax (per query column; partner lane is lane^16) --------
    float cm = -1e30f;
#pragma unroll
    for (int sub = 0; sub < 2; ++sub)
#pragma unroll
      for (int j = 0; j < 8; ++j) cm = fmaxf(cm, sv[sub][j]);
    cm = fmaxf(cm, __shfl_xor(cm, 16, 32));
    float mn = fmaxf(m, cm);
    float scale = __expf(m - mn);
    m = mn;
    l *= scale;
#pragma unroll
    for (int f = 0; f < 4; ++f)
#pragma unroll
      for (int j = 0; j < 8; ++j) acc[f][j] *= scale;

    float ps = 0.f;
    u32 pk0[4], pk1[4];
#pragma unroll
    for (int jj = 0; jj < 4; ++jj) {
      float a0 = __expf(sv[0][2 * jj] - m), a1 = __expf(sv[0][2 * jj + 1] - m);
      float b0 = __expf(sv[1][2 * jj] - m), b1 = __expf(sv[1][2 * jj + 1] - m);
      ps += a0 + a1 + b0 + b1;
      pk0[jj] = pack2h(a0, a1);
      pk1[jj] = pack2h(b0, b1);
    }
    l += ps + __shfl_xor(ps, 16, 32);

    // ---- build P^T B-fragment: half own regs, half partner (lane^16) -------
    u32 pp0[4], pp1[4];
#pragma unroll
    for (int p = 0; p < 4; ++p) {
      pp0[p] = __shfl_xor(pk0[p], 16, 32);
      pp1[p] = __shfl_xor(pk1[p], 16, 32);
    }
    union { v16h h; u32 u[8]; } bp;
#pragma unroll
    for (int p = 0; p < 4; ++p) {
      bp.u[p]     = hi ? pp1[p] : pk0[p];   // keys 16*hi + 0..7
      bp.u[4 + p] = hi ? pk1[p] : pp0[p];   // keys 16*hi + 8..15
    }

    // ---- out^T += V^T(tile f) · P^T ---------------------------------------
    v8h vf0[4], vf1[4];
#pragma unroll
    for (int f = 0; f < 4; ++f) {
      const half_t* vr = Vp + (size_t)(f * 16 + col) * T_ + kb + 16 * hi;
      vf0[f] = ((const v8h*)vr)[0];
      vf1[f] = ((const v8h*)(vr + 8))[0];
    }
#pragma unroll
    for (int f = 0; f < 4; ++f)
      acc[f] = WMMA(cat8(vf0[f], vf1[f]), bp.h, acc[f]);
  }

  const float inv = (l > 0.f) ? 1.0f / l : 0.f;
  const int bb = bh >> 4, hh = bh & 15;
  half_t* xr = Xo + ((size_t)bb * T_ + qbase + col) * D_ + hh * 64;
#pragma unroll
  for (int f = 0; f < 4; ++f) {
    v8h o;
#pragma unroll
    for (int j = 0; j < 8; ++j) o[j] = (half_t)(acc[f][j] * inv);
    *(v8h*)(xr + f * 16 + 8 * hi) = o;
  }
}

// ---------------------------------------------------------------------------
// Output projection: out(f32) = Xattn(f16) @ Wo^T + bo. Same structure as
// proj_featmajor but f32 store to d_out.
// ---------------------------------------------------------------------------
__global__ __launch_bounds__(256) void out_proj(
    const half_t* __restrict__ Xh,   // (NTOK, D) f16
    const half_t* __restrict__ Wh,   // (D, D) f16
    const float* __restrict__ bias,  // (D)
    float* __restrict__ Out)         // (NTOK, D) f32
{
  const int lane = threadIdx.x & 31;
  const int col  = lane & 15;
  const int hi   = lane >> 4;
  const int w    = blockIdx.x * 8 + (threadIdx.x >> 5);
  const int fbase   = (w & 63) * 16;
  const int tokbase = (w >> 6) * 64;

  v8f acc[4];
#pragma unroll
  for (int nt = 0; nt < 4; ++nt)
#pragma unroll
    for (int j = 0; j < 8; ++j) acc[nt][j] = 0.f;

  const half_t* wrow = Wh + (size_t)(fbase + col) * D_;
  const half_t* xrow[4];
#pragma unroll
  for (int nt = 0; nt < 4; ++nt)
    xrow[nt] = Xh + (size_t)(tokbase + nt * 16 + col) * D_ + 16 * hi;

  for (int kb = 0; kb < D_; kb += 32) {
    v8h a0 = *(const v8h*)(wrow + kb + 16 * hi);
    v8h a1 = *(const v8h*)(wrow + kb + 16 * hi + 8);
    v8h b0[4], b1[4];
#pragma unroll
    for (int nt = 0; nt < 4; ++nt) {
      b0[nt] = ((const v8h*)(xrow[nt] + kb))[0];
      b1[nt] = ((const v8h*)(xrow[nt] + kb))[1];
    }
    v16h a = cat8(a0, a1);
#pragma unroll
    for (int nt = 0; nt < 4; ++nt)
      acc[nt] = WMMA(a, cat8(b0[nt], b1[nt]), acc[nt]);
  }
  float bv[8];
#pragma unroll
  for (int r = 0; r < 8; ++r) bv[r] = bias[fbase + 8 * hi + r];
#pragma unroll
  for (int nt = 0; nt < 4; ++nt) {
    int tok = tokbase + nt * 16 + col;
    float* dst = Out + (size_t)tok * D_ + fbase + 8 * hi;
    v4f o0, o1;
#pragma unroll
    for (int r = 0; r < 4; ++r) { o0[r] = acc[nt][r] + bv[r]; o1[r] = acc[nt][4 + r] + bv[4 + r]; }
    ((v4f*)dst)[0] = o0;
    ((v4f*)dst)[1] = o1;
  }
}

// ---------------------------------------------------------------------------
extern "C" void kernel_launch(void* const* d_in, const int* in_sizes, int n_in,
                              void* d_out, int out_size, void* d_ws, size_t ws_size,
                              hipStream_t stream) {
  (void)in_sizes; (void)n_in; (void)out_size;
  const float* q  = (const float*)d_in[0];
  const float* k  = (const float*)d_in[1];
  const float* v  = (const float*)d_in[2];
  const unsigned char* mask = (const unsigned char*)d_in[3];
  const float* Wq = (const float*)d_in[4];  const float* bq = (const float*)d_in[5];
  const float* Wk = (const float*)d_in[6];  const float* bk = (const float*)d_in[7];
  const float* Wv = (const float*)d_in[8];  const float* bv = (const float*)d_in[9];
  const float* Wo = (const float*)d_in[10]; const float* bo = (const float*)d_in[11];
  float* out = (float*)d_out;

  // workspace carve-up (all f16), 256B aligned slices; total ~64 MB
  const size_t DD  = (size_t)D_ * D_;            // 1 Mi elements
  const size_t ACT = (size_t)NTOK_ * D_;         // 4 Mi elements
  const size_t QKV = (size_t)B_ * H_ * T_ * DK_; // 4 Mi elements
  char* ws = (char*)d_ws;
  size_t off = 0;
  auto carve = [&](size_t elems) {
    void* p = ws + off;
    off += (elems * sizeof(half_t) + 255) & ~(size_t)255;
    return (half_t*)p;
  };
  half_t* wq16 = carve(DD);
  half_t* wk16 = carve(DD);
  half_t* wv16 = carve(DD);
  half_t* wo16 = carve(DD);
  half_t* q16  = carve(ACT);
  half_t* k16  = carve(ACT);
  half_t* v16  = carve(ACT);
  half_t* Qh   = carve(QKV);
  half_t* Kh   = carve(QKV);
  half_t* Vth  = carve(QKV);
  half_t* Xa   = carve(ACT);
  if (ws_size < off) return;  // insufficient scratch (should not happen)

  dim3 blk(256);
  const int nw4 = (int)(DD / 4);
  const int na4 = (int)(ACT / 4);
  cvt_f32_f16<<<(nw4 + 255) / 256, blk, 0, stream>>>(Wq, wq16, nw4);
  cvt_f32_f16<<<(nw4 + 255) / 256, blk, 0, stream>>>(Wk, wk16, nw4);
  cvt_f32_f16<<<(nw4 + 255) / 256, blk, 0, stream>>>(Wv, wv16, nw4);
  cvt_f32_f16<<<(nw4 + 255) / 256, blk, 0, stream>>>(Wo, wo16, nw4);
  cvt_f32_f16<<<(na4 + 255) / 256, blk, 0, stream>>>(q, q16, na4);
  cvt_f32_f16<<<(na4 + 255) / 256, blk, 0, stream>>>(k, k16, na4);
  cvt_f32_f16<<<(na4 + 255) / 256, blk, 0, stream>>>(v, v16, na4);

  proj_featmajor   <<<512, blk, 0, stream>>>(q16, wq16, bq, Qh);
  proj_featmajor   <<<512, blk, 0, stream>>>(k16, wk16, bk, Kh);
  proj_v_transposed<<<512, blk, 0, stream>>>(v16, wv16, bv, Vth);
  flash_attn       <<<512, blk, 0, stream>>>(Qh, Kh, Vth, mask, Xa);
  out_proj         <<<512, blk, 0, stream>>>(Xa, wo16, bo, out);
}
